// PointSAModuleMSG_4209067950247
// MI455X (gfx1250) — matrix operand order repaired and checked by reading, not compiled, not run on hardware
//
#include <hip/hip_runtime.h>
#include <math.h>

// ---------------------------------------------------------------------------
// Problem constants (match reference)
// ---------------------------------------------------------------------------
#define BB 8
#define NN 8192
#define MM 2048
#define FEATC 64
#define OUTC 256          // 128 + 128 concatenated
#define EPSBN 1e-5f

typedef _Float16 half16 __attribute__((ext_vector_type(16)));
typedef _Float16 half8  __attribute__((ext_vector_type(8)));
typedef float    float8 __attribute__((ext_vector_type(8)));

// ---------------------------------------------------------------------------
// 1) Farthest point sampling: one block per batch, 2048 sequential steps.
//    dists live in LDS (32KB); two-level (shfl + LDS) argmax per step.
// ---------------------------------------------------------------------------
__global__ __launch_bounds__(1024)
void fps_kernel(const float* __restrict__ xyz,
                float* __restrict__ newxyz,     // (B,M,3)
                float* __restrict__ idx_out)    // (B,M) written as float
{
    __shared__ float dists[NN];
    __shared__ float wmax[32];
    __shared__ int   warg[32];
    __shared__ int   s_far;

    const int b = blockIdx.x;
    const int t = threadIdx.x;
    const float* px = xyz + (size_t)b * NN * 3;

    for (int n = t; n < NN; n += 1024) dists[n] = 1e10f;
    if (t == 0) s_far = 0;
    __syncthreads();

    for (int it = 0; it < MM; ++it) {
        const int far = s_far;
        const float cx = px[far * 3 + 0];
        const float cy = px[far * 3 + 1];
        const float cz = px[far * 3 + 2];
        if (t == 0) {
            idx_out[(size_t)b * MM + it] = (float)far;
            float* nx = newxyz + ((size_t)b * MM + it) * 3;
            nx[0] = cx; nx[1] = cy; nx[2] = cz;
        }
        float best = -1.0f;
        int   bestn = 0;
        for (int n = t; n < NN; n += 1024) {
            const float dx = px[3 * n + 0] - cx;
            const float dy = px[3 * n + 1] - cy;
            const float dz = px[3 * n + 2] - cz;
            const float d  = dx * dx + dy * dy + dz * dz;
            const float dm = fminf(dists[n], d);
            dists[n] = dm;
            if (dm > best) { best = dm; bestn = n; }
        }
        #pragma unroll
        for (int off = 16; off > 0; off >>= 1) {
            const float ob = __shfl_down(best, off);
            const int   on = __shfl_down(bestn, off);
            if (ob > best || (ob == best && on < bestn)) { best = ob; bestn = on; }
        }
        const int wid = t >> 5;
        if ((t & 31) == 0) { wmax[wid] = best; warg[wid] = bestn; }
        __syncthreads();
        if (t < 32) {
            best = wmax[t]; bestn = warg[t];
            #pragma unroll
            for (int off = 16; off > 0; off >>= 1) {
                const float ob = __shfl_down(best, off);
                const int   on = __shfl_down(bestn, off);
                if (ob > best || (ob == best && on < bestn)) { best = ob; bestn = on; }
            }
            if (t == 0) s_far = bestn;
        }
        __syncthreads();
    }
}

// ---------------------------------------------------------------------------
// 2) Feature transpose: (B,64,N) f32 -> (B*N, 64) f16 point-major rows.
// ---------------------------------------------------------------------------
__global__ void ftrans_kernel(const float* __restrict__ feats, _Float16* __restrict__ Fh)
{
    const int i = blockIdx.x * blockDim.x + threadIdx.x;
    if (i >= BB * NN * FEATC) return;
    const int c = i % FEATC;
    const int n = (i / FEATC) % NN;
    const int b = i / (FEATC * NN);
    Fh[i] = (_Float16)feats[((size_t)b * FEATC + c) * NN + n];
}

// ---------------------------------------------------------------------------
// 3) Ball query + group. POINT-MAJOR output: Xh[p][c], row stride 96 halfs
//    (rows 67..95 zero-padded => no bounds checks in the GEMM).
// ---------------------------------------------------------------------------
template <int K>
__global__ void ballgroup_kernel(const float* __restrict__ xyz,
                                 const _Float16* __restrict__ Fh,   // (B*N, 64)
                                 const float* __restrict__ newxyz,  // (B,M,3)
                                 _Float16* __restrict__ Xh,         // (P x 96)
                                 float r2)
{
    const int gid = blockIdx.x * blockDim.x + threadIdx.x;
    if (gid >= BB * MM) return;
    const int b = gid / MM;

    const float* nx = newxyz + (size_t)gid * 3;
    const float cx = nx[0], cy = nx[1], cz = nx[2];
    const float* px = xyz + (size_t)b * NN * 3;

    int idxs[K];
    int cnt = 0;
    for (int n = 0; n < NN && cnt < K; ++n) {
        const float dx = px[3 * n + 0] - cx;
        const float dy = px[3 * n + 1] - cy;
        const float dz = px[3 * n + 2] - cz;
        if (dx * dx + dy * dy + dz * dz < r2) idxs[cnt++] = n;
    }
    if (cnt == 0) idxs[cnt++] = 0;
    for (int k = cnt; k < K; ++k) idxs[k] = idxs[0];

    for (int k = 0; k < K; ++k) {
        const int i = idxs[k];
        _Float16* row = Xh + ((size_t)gid * K + k) * 96;
        row[0] = (_Float16)(px[3 * i + 0] - cx);
        row[1] = (_Float16)(px[3 * i + 1] - cy);
        row[2] = (_Float16)(px[3 * i + 2] - cz);
        const _Float16* fr = Fh + ((size_t)b * NN + i) * FEATC;
        #pragma unroll
        for (int c8 = 0; c8 < FEATC; c8 += 8) {
            const half8 f = *(const half8*)(fr + c8);   // 16B aligned
            #pragma unroll
            for (int e = 0; e < 8; ++e) row[3 + c8 + e] = f[e];
        }
        #pragma unroll
        for (int c = 67; c < 96; ++c) row[c] = (_Float16)0.0f;
    }
}

// ---------------------------------------------------------------------------
// 4) Weight prep: f32 (Cout,Cin) -> f16 (Cout,CinPad) zero-padded.
// ---------------------------------------------------------------------------
__global__ void wprep_kernel(const float* __restrict__ W, _Float16* __restrict__ Wh,
                             int Cin, int CinPad, int Cout)
{
    const int i = blockIdx.x * blockDim.x + threadIdx.x;
    if (i >= Cout * CinPad) return;
    const int c = i / CinPad, k = i % CinPad;
    Wh[i] = (k < Cin) ? (_Float16)W[(size_t)c * Cin + k] : (_Float16)0.0f;
}

// ---------------------------------------------------------------------------
// 5) WMMA GEMM, point-major: Y[p][co] = sum_k Wh[co][k] * X[p][k].
//    Each wave loads its B (activation) fragments ONCE, then loops over all
//    Cout tiles re-loading only the L1-resident weights -> X read exactly
//    once from memory. Inner loop: 2 b128-load clauses per k-step, chained
//    WMMAs, one b128 store. No LDS, no atomics in the hot path.
// ---------------------------------------------------------------------------
template <int CINPAD>
__global__ __launch_bounds__(128)
void wmma_gemm(const _Float16* __restrict__ X,   // (P x CINPAD) f16
               const _Float16* __restrict__ Wh,  // (Cout x CINPAD) f16
               _Float16*       __restrict__ Y,   // (P x Cout) f16
               int Cout, int P)
{
    constexpr int KSTEPS = CINPAD / 32;
    const int lane = threadIdx.x & 31;
    const int wv   = threadIdx.x >> 5;
    const int n0   = (blockIdx.x * 4 + wv) * 16;
    const int hi   = lane >> 4;       // lane-half: 0 or 1
    const int l15  = lane & 15;

    // B fragments (32x16 per k-step): lane col = l15; elems -> K contiguous.
    const _Float16* xrow = X + (size_t)(n0 + l15) * CINPAD;
    half16 bfrag[KSTEPS];
    #pragma unroll
    for (int s = 0; s < KSTEPS; ++s) {
        const half8 b0 = *(const half8*)(xrow + s * 32 + (hi << 4));
        const half8 b1 = *(const half8*)(xrow + s * 32 + (hi << 4) + 8);
        #pragma unroll
        for (int e = 0; e < 8; ++e) { bfrag[s][e] = b0[e]; bfrag[s][e + 8] = b1[e]; }
    }

    _Float16* yrow = Y + (size_t)(n0 + l15) * Cout + (hi << 3);

    for (int m0 = 0; m0 < Cout; m0 += 16) {
        const _Float16* wrow = Wh + (size_t)(m0 + l15) * CINPAD;
        float8 acc = {};
        #pragma unroll
        for (int s = 0; s < KSTEPS; ++s) {
            // A (16x32): lane row = l15; elems 0..15 -> K=+16*hi (contiguous)
            const half8 a0 = *(const half8*)(wrow + s * 32 + (hi << 4));
            const half8 a1 = *(const half8*)(wrow + s * 32 + (hi << 4) + 8);
            half16 a;
            #pragma unroll
            for (int e = 0; e < 8; ++e) { a[e] = a0[e]; a[e + 8] = a1[e]; }
            acc = __builtin_amdgcn_wmma_f32_16x16x32_f16(
                false, a, false, bfrag[s], (short)0, acc, false, false);
        }
        // D: VGPR d, lane L -> M = m0 + d + 8*hi (contiguous over d), N = n0 + l15
        half8 dv;
        #pragma unroll
        for (int d = 0; d < 8; ++d) dv[d] = (_Float16)acc[d];
        *(half8*)(yrow + m0) = dv;
    }
}

// ---------------------------------------------------------------------------
// 6) Per-channel sum / sum^2 over Y: block = (8-channel group) x point-slab.
//    Register accumulation from b128 row loads, one shfl reduce per wave,
//    O(Cout * slabs) atomics total.
// ---------------------------------------------------------------------------
__global__ __launch_bounds__(256)
void stats_kernel(const _Float16* __restrict__ Y,   // (P x Cout)
                  float* __restrict__ sum, float* __restrict__ sumsq,
                  int Cout, int P)
{
    const int cg   = blockIdx.x;          // channel group: 8 channels
    const int slab = blockIdx.y;
    const int nsl  = gridDim.y;
    const int pps  = P / nsl;             // points per slab (P divisible)
    const int t    = threadIdx.x;

    float s[8], q[8];
    #pragma unroll
    for (int e = 0; e < 8; ++e) { s[e] = 0.0f; q[e] = 0.0f; }

    const int p0 = slab * pps;
    for (int p = p0 + t; p < p0 + pps; p += 256) {
        const half8 v = *(const half8*)(Y + (size_t)p * Cout + cg * 8);
        #pragma unroll
        for (int e = 0; e < 8; ++e) {
            const float f = (float)v[e];
            s[e] += f;
            q[e] += f * f;
        }
    }
    #pragma unroll
    for (int e = 0; e < 8; ++e) {
        #pragma unroll
        for (int off = 16; off > 0; off >>= 1) {
            s[e] += __shfl_down(s[e], off);
            q[e] += __shfl_down(q[e], off);
        }
    }
    if ((t & 31) == 0) {
        #pragma unroll
        for (int e = 0; e < 8; ++e) {
            atomicAdd(&sum[cg * 8 + e],   s[e]);
            atomicAdd(&sumsq[cg * 8 + e], q[e]);
        }
    }
}

// ---------------------------------------------------------------------------
// 7) In-place batch-norm + affine + ReLU, one b128 load/store per 8 channels.
// ---------------------------------------------------------------------------
__global__ void norm_relu_kernel(_Float16* __restrict__ Y,   // (P x Cout)
                                 const float* __restrict__ sum,
                                 const float* __restrict__ sumsq,
                                 const float* __restrict__ gamma,
                                 const float* __restrict__ beta,
                                 int Cout, int P)
{
    const size_t i    = (size_t)blockIdx.x * blockDim.x + threadIdx.x;
    const size_t tot8 = (size_t)Cout * P / 8;
    if (i >= tot8) return;
    const size_t base = i * 8;
    const int    c0   = (int)(base % Cout);
    const float  invP = 1.0f / (float)P;

    half8 v = *(const half8*)(Y + base);
    half8 r;
    #pragma unroll
    for (int e = 0; e < 8; ++e) {
        const int   c    = c0 + e;
        const float mean = sum[c] * invP;
        const float var  = sumsq[c] * invP - mean * mean;
        float x = (float)v[e];
        x = (x - mean) * rsqrtf(var + EPSBN) * gamma[c] + beta[c];
        r[e] = (_Float16)(x > 0.0f ? x : 0.0f);
    }
    *(half8*)(Y + base) = r;
}

// ---------------------------------------------------------------------------
// 8) Max over K: thread per (point, 8-channel group), b128 loads per k.
//    Writes (B,256,M) channel-major output at offset co_base.
// ---------------------------------------------------------------------------
__global__ void maxpool_kernel(const _Float16* __restrict__ X,  // (P x 128)
                               float* __restrict__ out,         // (B,256,M)
                               int Kk, int co_base)
{
    const int i = blockIdx.x * blockDim.x + threadIdx.x;
    if (i >= BB * MM * 16) return;               // 16 groups of 8 channels
    const int cg = i % 16;
    const int m  = (i / 16) % MM;
    const int b  = i / (16 * MM);

    const _Float16* base = X + ((size_t)(b * MM + m) * Kk) * 128 + cg * 8;
    float best[8];
    #pragma unroll
    for (int e = 0; e < 8; ++e) best[e] = 0.0f;  // post-ReLU values >= 0
    for (int k = 0; k < Kk; ++k) {
        const half8 v = *(const half8*)(base + (size_t)k * 128);
        #pragma unroll
        for (int e = 0; e < 8; ++e) best[e] = fmaxf(best[e], (float)v[e]);
    }
    float* o = out + ((size_t)b * OUTC + co_base + cg * 8) * MM + m;
    #pragma unroll
    for (int e = 0; e < 8; ++e) o[(size_t)e * MM] = best[e];
}

__global__ void zero_kernel(float* p, int n)
{
    const int i = blockIdx.x * blockDim.x + threadIdx.x;
    if (i < n) p[i] = 0.0f;
}

// ---------------------------------------------------------------------------
// Host orchestration
// ---------------------------------------------------------------------------
extern "C" void kernel_launch(void* const* d_in, const int* in_sizes, int n_in,
                              void* d_out, int out_size, void* d_ws, size_t ws_size,
                              hipStream_t stream)
{
    const float* xyz   = (const float*)d_in[0];   // (B,N,3)
    const float* feats = (const float*)d_in[1];   // (B,64,N)

    float* out      = (float*)d_out;
    float* new_xyz  = out;                                      // B*M*3
    float* new_feat = out + (size_t)BB * MM * 3;                // B*256*M
    float* idx_out  = new_feat + (size_t)BB * OUTC * MM;        // B*M (as float)

    // Workspace: two f16 ping-pong buffers (Pmax x 128) + stats + Wh + Fh.
    const size_t Pmax  = (size_t)BB * MM * 32;                  // 524288
    const size_t bufsz = Pmax * 128 * sizeof(_Float16);
    _Float16* bufA  = (_Float16*)d_ws;
    _Float16* bufB  = (_Float16*)((char*)d_ws + bufsz);
    float*    stats = (float*)((char*)d_ws + 2 * bufsz);
    float*    ssum  = stats;
    float*    ssq   = stats + 128;
    _Float16* Wh    = (_Float16*)((char*)d_ws + 2 * bufsz + 1024);         // <=24KB
    _Float16* Fh    = (_Float16*)((char*)d_ws + 2 * bufsz + 1024 + 65536); // 8MB

    // --- FPS (serial-chain bottleneck; one big block per batch) ---
    fps_kernel<<<BB, 1024, 0, stream>>>(xyz, new_xyz, idx_out);

    // --- one-time feature transpose to point-major f16 ---
    ftrans_kernel<<<(BB * NN * FEATC + 255) / 256, 256, 0, stream>>>(feats, Fh);

    const float radii[2] = {0.2f, 0.4f};
    const int   ks[2]    = {16, 32};
    const int   cins[2][3]  = {{67, 64, 64}, {67, 64, 96}};
    const int   couts[2][3] = {{64, 64, 128}, {64, 96, 128}};

    for (int br = 0; br < 2; ++br) {
        const int K = ks[br];
        const int P = BB * MM * K;
        const float r2 = radii[br] * radii[br];

        if (K == 16)
            ballgroup_kernel<16><<<(BB * MM + 127) / 128, 128, 0, stream>>>(
                xyz, Fh, new_xyz, bufA, r2);
        else
            ballgroup_kernel<32><<<(BB * MM + 127) / 128, 128, 0, stream>>>(
                xyz, Fh, new_xyz, bufA, r2);

        _Float16* cur = bufA;
        _Float16* oth = bufB;
        for (int l = 0; l < 3; ++l) {
            const int cin    = cins[br][l];
            const int cinPad = (cin + 31) & ~31;     // 96 or 64
            const int cout   = couts[br][l];
            const float* w = (const float*)d_in[2 + br * 9 + l * 3 + 0];
            const float* g = (const float*)d_in[2 + br * 9 + l * 3 + 1];
            const float* b = (const float*)d_in[2 + br * 9 + l * 3 + 2];

            wprep_kernel<<<(cout * cinPad + 255) / 256, 256, 0, stream>>>(
                w, Wh, cin, cinPad, cout);
            zero_kernel<<<1, 256, 0, stream>>>(stats, 256);

            dim3 grid(P / 64);                       // 4 waves/block, 16 pts each
            if (cinPad == 96)
                wmma_gemm<96><<<grid, 128, 0, stream>>>(cur, Wh, oth, cout, P);
            else
                wmma_gemm<64><<<grid, 128, 0, stream>>>(cur, Wh, oth, cout, P);

            dim3 sgrid(cout / 8, 256);               // 8-ch groups x point slabs
            stats_kernel<<<sgrid, 256, 0, stream>>>(oth, ssum, ssq, cout, P);

            const size_t tot8 = (size_t)cout * P / 8;
            norm_relu_kernel<<<(unsigned)((tot8 + 255) / 256), 256, 0, stream>>>(
                oth, ssum, ssq, g, b, cout, P);

            _Float16* t = cur; cur = oth; oth = t;
        }
        const int tot = BB * MM * 16;
        maxpool_kernel<<<(tot + 255) / 256, 256, 0, stream>>>(
            cur, new_feat, K, br * 128);
    }
    (void)in_sizes; (void)n_in; (void)out_size; (void)ws_size;
}
